// T5Attention_47339129537051
// MI455X (gfx1250) — matrix operand order repaired and checked by reading, not compiled, not run on hardware
//
#include <hip/hip_runtime.h>
#include <hip/hip_bf16.h>
#include <cstdint>

typedef __attribute__((ext_vector_type(16))) __bf16       v16bf;
typedef __attribute__((ext_vector_type(8)))  float        v8f;
typedef __attribute__((ext_vector_type(4)))  float        v4f;
typedef __attribute__((ext_vector_type(4)))  unsigned int v4u;
typedef __attribute__((ext_vector_type(2)))  unsigned int v2u;

union FragBF { v16bf v; v4u q[2]; unsigned short u[16]; };
union FragF  { v8f   v; float f[8]; };

__device__ __forceinline__ unsigned short f2bf(float f) {
  unsigned u = __float_as_uint(f);
  unsigned r = u + 0x7FFFu + ((u >> 16) & 1u);   // round-to-nearest-even
  return (unsigned short)(r >> 16);
}

// ---------------------------------------------------------------------------
// Tiled WMMA GEMM: C[M,N] = A[M,K] * B[K,N]; B is fp32 (weights).
// A is fp32 (A_BF16=false) or bf16 (true); C is bf16 (OUT_F32=false) or fp32.
// Block 256 threads = 8 waves; tile 128x128, K-step 32 (one WMMA K).
// Waves arranged 4(M) x 2(N); each wave owns 2x4 = 8 16x16 WMMA tiles.
// ---------------------------------------------------------------------------
template<bool A_BF16, bool OUT_F32>
__global__ __launch_bounds__(256)
void gemm_wmma(const void* __restrict__ Ap, const float* __restrict__ Bp,
               void* __restrict__ Cp, int M, int N, int K)
{
  __shared__ unsigned short As[128 * 40];  // [row][k] pad 32->40
  __shared__ unsigned short Bt[128 * 40];  // [n][k] (B transposed) pad 32->40

  const int tid  = threadIdx.x;
  const int lane = tid & 31;
  const int rsel = lane >> 4;          // 0: lanes 0-15, 1: lanes 16-31
  const int wid  = tid >> 5;
  const int wm   = wid & 3;            // wave row group (32 rows each)
  const int wn   = wid >> 2;           // wave col group (64 cols each)
  const int bM   = blockIdx.y * 128;
  const int bN   = blockIdx.x * 128;

  FragF acc[2][4] = {};

  for (int k0 = 0; k0 < K; k0 += 32) {
    // ---- stage A tile (128x32) into LDS as bf16 ----
    if constexpr (A_BF16) {
      const unsigned short* A = (const unsigned short*)Ap;
#pragma unroll
      for (int i = 0; i < 2; ++i) {
        int c = tid + i * 256;                 // 512 chunks of 8 bf16
        int row = c >> 2, col = (c & 3) * 8;
        v4u val = *(const v4u*)(A + (size_t)(bM + row) * K + k0 + col);
        *(v4u*)(&As[row * 40 + col]) = val;
      }
    } else {
      const float* A = (const float*)Ap;
#pragma unroll
      for (int i = 0; i < 4; ++i) {
        int c = tid + i * 256;                 // 1024 chunks of 4 f32
        int row = c >> 3, col = (c & 7) * 4;
        v4f val = *(const v4f*)(A + (size_t)(bM + row) * K + k0 + col);
        v2u pk;
        pk.x = (unsigned)f2bf(val.x) | ((unsigned)f2bf(val.y) << 16);
        pk.y = (unsigned)f2bf(val.z) | ((unsigned)f2bf(val.w) << 16);
        *(v2u*)(&As[row * 40 + col]) = pk;
      }
    }
    // ---- stage B tile (32x128) transposed into LDS as bf16 ----
#pragma unroll
    for (int i = 0; i < 4; ++i) {
      int c = tid + i * 256;                   // 1024 chunks of 4 f32
      int row = c >> 5, col = (c & 31) * 4;    // row = k, col = n
      v4f val = *(const v4f*)(Bp + (size_t)(k0 + row) * N + bN + col);
      Bt[(col + 0) * 40 + row] = f2bf(val.x);
      Bt[(col + 1) * 40 + row] = f2bf(val.y);
      Bt[(col + 2) * 40 + row] = f2bf(val.z);
      Bt[(col + 3) * 40 + row] = f2bf(val.w);
    }

    // ---- prefetch next K-step tiles into cache while we compute this one ----
    if (k0 + 32 < K) {
      int k1 = k0 + 32;
      if constexpr (A_BF16) {
        const unsigned short* A = (const unsigned short*)Ap;
        int row = tid >> 1, col = (tid & 1) * 16;    // 2 cachelines-ish per row
        __builtin_prefetch(A + (size_t)(bM + row) * K + k1 + col, 0, 1);
      } else {
        const float* A = (const float*)Ap;
        int row = tid >> 1, col = (tid & 1) * 16;
        __builtin_prefetch(A + (size_t)(bM + row) * K + k1 + col, 0, 1);
      }
      {
        int row = tid >> 3, col = (tid & 7) * 16;    // 32 rows x 128 f32
        __builtin_prefetch(Bp + (size_t)(k1 + row) * N + bN + col, 0, 1);
      }
    }
    __syncthreads();

    // ---- fragments + WMMA ----
    FragBF af[2], bfr[4];
#pragma unroll
    for (int tm = 0; tm < 2; ++tm) {
      const unsigned short* p = &As[(wm * 32 + tm * 16 + (lane & 15)) * 40 + rsel * 8];
      af[tm].q[0] = *(const v4u*)p;            // elems 0-7 : k = rsel*8 + 0..7
      af[tm].q[1] = *(const v4u*)(p + 16);     // elems 8-15: k = 16 + rsel*8 + 0..7
    }
#pragma unroll
    for (int tn = 0; tn < 4; ++tn) {
      const unsigned short* p = &Bt[(wn * 64 + tn * 16 + (lane & 15)) * 40 + rsel * 8];
      bfr[tn].q[0] = *(const v4u*)p;           // elems 0-7 : k = rsel*8 + 0..7
      bfr[tn].q[1] = *(const v4u*)(p + 16);
    }
#pragma unroll
    for (int tm = 0; tm < 2; ++tm)
#pragma unroll
      for (int tn = 0; tn < 4; ++tn)
        acc[tm][tn].v = __builtin_amdgcn_wmma_f32_16x16x32_bf16(
            false, af[tm].v, false, bfr[tn].v, (short)0, acc[tm][tn].v, false, false);
    __syncthreads();
  }

  // ---- epilogue ----
#pragma unroll
  for (int tm = 0; tm < 2; ++tm)
#pragma unroll
    for (int tn = 0; tn < 4; ++tn)
#pragma unroll
      for (int r = 0; r < 8; ++r) {
        int gm = bM + wm * 32 + tm * 16 + rsel * 8 + r;
        int gn = bN + wn * 64 + tn * 16 + (lane & 15);
        if constexpr (OUT_F32)
          ((float*)Cp)[(size_t)gm * N + gn] = acc[tm][tn].f[r];
        else
          ((unsigned short*)Cp)[(size_t)gm * N + gn] = f2bf(acc[tm][tn].f[r]);
      }
}

// ---------------------------------------------------------------------------
// Fused flash attention with T5 relative-position bias (encoder, bidirectional).
// Grid (T/64, H, B); block 128 = 4 waves; wave owns 16 query rows; key blocks
// of 64. QK^T and P*V via v_wmma_f32_16x16x32_bf16, online softmax in fp32.
// ---------------------------------------------------------------------------
__global__ __launch_bounds__(128)
void flash_t5(const unsigned short* __restrict__ Qb,
              const unsigned short* __restrict__ Kb,
              const unsigned short* __restrict__ Vb,
              const float* __restrict__ rel_bias,
              unsigned short* __restrict__ Ob)
{
  constexpr int T = 2048, D = 1024, DH = 64, H = 16;
  __shared__ unsigned short Qs [64 * 72];  // [qrow][dh]
  __shared__ unsigned short Ks [64 * 72];  // [key][dh]
  __shared__ unsigned short Vst[64 * 72];  // [dh][key]  (V transposed)
  __shared__ unsigned short Ps [64 * 72];  // per-wave 16 rows: [qrow][key]
  __shared__ float rb[32];

  const int tid  = threadIdx.x;
  const int lane = tid & 31;
  const int rsel = lane >> 4;
  const int wave = tid >> 5;
  const int h  = blockIdx.y;
  const int b  = blockIdx.z;
  const int t0 = blockIdx.x * 64;

  if (tid < 32) rb[tid] = rel_bias[tid * H + h];

  // load 64x64 bf16 Q block
#pragma unroll
  for (int i = 0; i < 4; ++i) {
    int c = tid + i * 128;
    int row = c >> 3, col = (c & 7) * 8;
    v4u val = *(const v4u*)(Qb + (size_t)(b * T + t0 + row) * D + h * DH + col);
    *(v4u*)(&Qs[row * 72 + col]) = val;
  }
  __syncthreads();

  FragBF qf[2];
#pragma unroll
  for (int w = 0; w < 2; ++w) {
    const unsigned short* p = &Qs[(wave * 16 + (lane & 15)) * 72 + w * 32 + rsel * 8];
    qf[w].q[0] = *(const v4u*)p;
    qf[w].q[1] = *(const v4u*)(p + 16);
  }

  float m_s[8], l_s[8];
  FragF o[4] = {};
#pragma unroll
  for (int r = 0; r < 8; ++r) { m_s[r] = -1e30f; l_s[r] = 0.f; }

  for (int s0 = 0; s0 < T; s0 += 64) {
    __syncthreads();
    // stage K (row-major) and V (transposed) blocks
#pragma unroll
    for (int i = 0; i < 4; ++i) {
      int c = tid + i * 128;
      int row = c >> 3, col = (c & 7) * 8;
      v4u kv = *(const v4u*)(Kb + (size_t)(b * T + s0 + row) * D + h * DH + col);
      *(v4u*)(&Ks[row * 72 + col]) = kv;
      v4u vv = *(const v4u*)(Vb + (size_t)(b * T + s0 + row) * D + h * DH + col);
      const unsigned short* vs = (const unsigned short*)&vv;
#pragma unroll
      for (int j = 0; j < 8; ++j) Vst[(col + j) * 72 + row] = vs[j];
    }
    // prefetch next key block (K and V rows) while this one is computed
    if (s0 + 64 < T) {
      int row = tid >> 1, col = (tid & 1) * 32;   // 64 rows x 64 bf16 = 2 x 64B
      const unsigned short* pk =
          Kb + (size_t)(b * T + s0 + 64 + row) * D + h * DH + col;
      const unsigned short* pv =
          Vb + (size_t)(b * T + s0 + 64 + row) * D + h * DH + col;
      __builtin_prefetch(pk, 0, 1);
      __builtin_prefetch(pv, 0, 1);
    }
    __syncthreads();

    // ---- S = Q K^T + bias ----
    float sv[4][8];
#pragma unroll
    for (int kt = 0; kt < 4; ++kt) {
      FragF accs = {};
      FragBF kf;
#pragma unroll
      for (int w = 0; w < 2; ++w) {
        const unsigned short* p = &Ks[(kt * 16 + (lane & 15)) * 72 + w * 32 + rsel * 8];
        kf.q[0] = *(const v4u*)p;
        kf.q[1] = *(const v4u*)(p + 16);
        accs.v = __builtin_amdgcn_wmma_f32_16x16x32_bf16(
            false, qf[w].v, false, kf.v, (short)0, accs.v, false, false);
      }
      int keyg = s0 + kt * 16 + (lane & 15);
#pragma unroll
      for (int r = 0; r < 8; ++r) {
        int qg = t0 + wave * 16 + rsel * 8 + r;
        int rp = keyg - qg;                       // mem - ctx
        int bkt = rp > 0 ? 16 : 0;
        int a = rp < 0 ? -rp : rp;
        int idx;
        if (a < 8) idx = a;
        else { idx = 33 - __clz(a * a); idx = idx > 15 ? 15 : idx; }
        sv[kt][r] = accs.f[r] + rb[bkt + idx];
      }
    }

    // ---- online softmax update (row reductions across 16-lane halves) ----
#pragma unroll
    for (int r = 0; r < 8; ++r) {
      float mx = fmaxf(fmaxf(sv[0][r], sv[1][r]), fmaxf(sv[2][r], sv[3][r]));
#pragma unroll
      for (int off = 8; off >= 1; off >>= 1) mx = fmaxf(mx, __shfl_xor(mx, off, 32));
      float mnew  = fmaxf(m_s[r], mx);
      float alpha = __expf(m_s[r] - mnew);
      float psum = 0.f;
#pragma unroll
      for (int kt = 0; kt < 4; ++kt) {
        float p = __expf(sv[kt][r] - mnew);
        sv[kt][r] = p;
        psum += p;
      }
#pragma unroll
      for (int off = 8; off >= 1; off >>= 1) psum += __shfl_xor(psum, off, 32);
      l_s[r] = l_s[r] * alpha + psum;
      m_s[r] = mnew;
#pragma unroll
      for (int ot = 0; ot < 4; ++ot) o[ot].f[r] *= alpha;
    }

    // ---- transpose P (C-layout -> A-layout) through per-wave LDS ----
#pragma unroll
    for (int kt = 0; kt < 4; ++kt)
#pragma unroll
      for (int r = 0; r < 8; ++r)
        Ps[(wave * 16 + rsel * 8 + r) * 72 + kt * 16 + (lane & 15)] = f2bf(sv[kt][r]);
    asm volatile("s_wait_dscnt 0" ::: "memory");   // same-wave LDS RAW fence

    FragBF pf[2];
#pragma unroll
    for (int w = 0; w < 2; ++w) {
      const unsigned short* p = &Ps[(wave * 16 + (lane & 15)) * 72 + w * 32 + rsel * 8];
      pf[w].q[0] = *(const v4u*)p;
      pf[w].q[1] = *(const v4u*)(p + 16);
    }

    // ---- O += P * V ----
#pragma unroll
    for (int ot = 0; ot < 4; ++ot) {
      FragBF vf;
#pragma unroll
      for (int w = 0; w < 2; ++w) {
        const unsigned short* p = &Vst[(ot * 16 + (lane & 15)) * 72 + w * 32 + rsel * 8];
        vf.q[0] = *(const v4u*)p;
        vf.q[1] = *(const v4u*)(p + 16);
        o[ot].v = __builtin_amdgcn_wmma_f32_16x16x32_bf16(
            false, pf[w].v, false, vf.v, (short)0, o[ot].v, false, false);
      }
    }
  }

  // ---- normalize and store bf16 [B,T,H,dh] ----
#pragma unroll
  for (int ot = 0; ot < 4; ++ot)
#pragma unroll
    for (int r = 0; r < 8; ++r) {
      int qg  = t0 + wave * 16 + rsel * 8 + r;
      int col = h * DH + ot * 16 + (lane & 15);
      Ob[(size_t)(b * T + qg) * D + col] = f2bf(o[ot].f[r] / l_s[r]);
    }
}

// ---------------------------------------------------------------------------
extern "C" void kernel_launch(void* const* d_in, const int* in_sizes, int n_in,
                              void* d_out, int out_size, void* d_ws, size_t ws_size,
                              hipStream_t stream) {
  (void)in_sizes; (void)n_in; (void)out_size; (void)ws_size;
  const float* q_in  = (const float*)d_in[0];   // [4,2048,1024]
  const float* kv_in = (const float*)d_in[1];   // [4,2048,1024]
  const float* wq    = (const float*)d_in[2];   // [1024,1024]
  const float* wk    = (const float*)d_in[3];
  const float* wv    = (const float*)d_in[4];
  const float* wo    = (const float*)d_in[5];
  const float* rbias = (const float*)d_in[6];   // [32,16]
  float* out = (float*)d_out;

  const int M = 4 * 2048, N = 1024, K = 1024;
  const size_t SZ = (size_t)M * N * sizeof(unsigned short); // 16 MiB per bf16 buffer
  unsigned char* ws = (unsigned char*)d_ws;
  unsigned short* Qb = (unsigned short*)(ws + 0 * SZ);
  unsigned short* Kb = (unsigned short*)(ws + 1 * SZ);
  unsigned short* Vb = (unsigned short*)(ws + 2 * SZ);
  unsigned short* Ob = (unsigned short*)(ws + 3 * SZ);

  dim3 gg(N / 128, M / 128);   // (8, 64)
  gemm_wmma<false, false><<<gg, 256, 0, stream>>>(q_in,  wq, Qb, M, N, K);
  gemm_wmma<false, false><<<gg, 256, 0, stream>>>(kv_in, wk, Kb, M, N, K);
  gemm_wmma<false, false><<<gg, 256, 0, stream>>>(kv_in, wv, Vb, M, N, K);

  flash_t5<<<dim3(2048 / 64, 16, 4), 128, 0, stream>>>(Qb, Kb, Vb, rbias, Ob);

  gemm_wmma<true, true><<<gg, 256, 0, stream>>>(Ob, wo, out, M, N, K);
}
